// BitLinear_14998025797774
// MI455X (gfx1250) — compile-verified, matrix operand order checked by reading, not run on hardware
//
#include <hip/hip_runtime.h>

#define K_DIM 4096
#define N_DIM 4096
#define BM 128
#define BN 128
#define BK 32
#define PAD 8
#define LDSROW (BK + PAD)   // 40 halfs = 80 B row stride; 16B aligned, bank-conflict free

#if defined(__has_builtin)
#if __has_builtin(__builtin_amdgcn_global_load_async_to_lds_b128) && \
    __has_builtin(__builtin_amdgcn_s_wait_asynccnt)
#define USE_ASYNC_LDS 1
#endif
#endif
#ifndef USE_ASYNC_LDS
#define USE_ASYNC_LDS 0
#endif

typedef int v4i __attribute__((ext_vector_type(4)));

// addrspace casts via integer round-trip (generic LDS ptr low 32 bits = LDS offset)
#define AS1P(p) ((__attribute__((address_space(1))) v4i*)(unsigned long long)(p))
#define AS3P(p) ((__attribute__((address_space(3))) v4i*)(unsigned int)(unsigned long long)(p))

typedef __bf16 v16bf __attribute__((ext_vector_type(16)));
typedef __bf16 v8bf  __attribute__((ext_vector_type(8)));
typedef float  v8f   __attribute__((ext_vector_type(8)));

union Frag { v16bf v; v8bf h[2]; };

__device__ __forceinline__ unsigned int f2bf(float f) {
  unsigned int u = __float_as_uint(f);
  return (u + 0x7FFFu + ((u >> 16) & 1u)) >> 16;   // RNE bf16
}

// ---------------------------------------------------------------------------
// Kernel 1: RMSNorm one 4096-wide row, emit bf16 into workspace.
// ---------------------------------------------------------------------------
__global__ void __launch_bounds__(256)
rmsnorm_bf16_kernel(const float* __restrict__ x,
                    const float* __restrict__ nw,
                    unsigned short* __restrict__ xn) {
  const int row  = blockIdx.x;
  const int tid  = threadIdx.x;
  const int base = tid * 16;
  const float* xr = x + (size_t)row * K_DIM;

  float vals[16];
  float ss = 0.f;
#pragma unroll
  for (int i = 0; i < 4; ++i) {
    float4 v = *(const float4*)(xr + base + i * 4);
    vals[4 * i + 0] = v.x; vals[4 * i + 1] = v.y;
    vals[4 * i + 2] = v.z; vals[4 * i + 3] = v.w;
    ss += v.x * v.x + v.y * v.y + v.z * v.z + v.w * v.w;
  }
#pragma unroll
  for (int off = 16; off > 0; off >>= 1) ss += __shfl_xor(ss, off, 32);
  __shared__ float wss[8];
  if ((tid & 31) == 0) wss[tid >> 5] = ss;
  __syncthreads();
  float tot = 0.f;
#pragma unroll
  for (int i = 0; i < 8; ++i) tot += wss[i];

  const float scale = rsqrtf(tot / (float)K_DIM + 1e-6f);

  float nws[16];
#pragma unroll
  for (int i = 0; i < 4; ++i) {
    float4 w = *(const float4*)(nw + base + i * 4);
    nws[4 * i + 0] = w.x; nws[4 * i + 1] = w.y;
    nws[4 * i + 2] = w.z; nws[4 * i + 3] = w.w;
  }

  unsigned int packed[8];
#pragma unroll
  for (int j = 0; j < 8; ++j) {
    float a = vals[2 * j]     * scale * nws[2 * j];
    float b = vals[2 * j + 1] * scale * nws[2 * j + 1];
    packed[j] = f2bf(a) | (f2bf(b) << 16);
  }
  uint4* dst = (uint4*)(xn + (size_t)row * K_DIM + base);
  dst[0] = make_uint4(packed[0], packed[1], packed[2], packed[3]);
  dst[1] = make_uint4(packed[4], packed[5], packed[6], packed[7]);
}

// ---------------------------------------------------------------------------
// Kernel 2: one-shot fp32 -> bf16 weight conversion (ternary values exact).
// ---------------------------------------------------------------------------
__global__ void __launch_bounds__(256)
w_to_bf16_kernel(const float* __restrict__ wq, unsigned short* __restrict__ wb) {
  const size_t i = ((size_t)blockIdx.x * 256 + threadIdx.x) * 8;
  float4 f0 = *(const float4*)(wq + i);
  float4 f1 = *(const float4*)(wq + i + 4);
  uint4 o;
  o.x = f2bf(f0.x) | (f2bf(f0.y) << 16);
  o.y = f2bf(f0.z) | (f2bf(f0.w) << 16);
  o.z = f2bf(f1.x) | (f2bf(f1.y) << 16);
  o.w = f2bf(f1.z) | (f2bf(f1.w) << 16);
  *(uint4*)(wb + i) = o;
}

// ---------------------------------------------------------------------------
// Kernel 3: y[m,n] = (sum_k xn[m,k]*w[n,k]) * gamma[n]
// 128x128 block, BK=32, double-buffered LDS, branchless 2x-unrolled pipeline
// with compile-time buffer indices. 8 waves, each 64x32 out-tile =
// 8x v_wmma_f32_16x16x32_bf16 per K-step. Async global->LDS when available.
// ---------------------------------------------------------------------------
template <bool WBF16>
__global__ void __launch_bounds__(256)
bitlinear_wmma_kernel(const unsigned short* __restrict__ xnb,
                      const unsigned short* __restrict__ wb,
                      const float* __restrict__ wq,
                      const float* __restrict__ gamma,
                      float* __restrict__ out) {
  __shared__ __align__(16) unsigned short sA[2][BM][LDSROW];
  __shared__ __align__(16) unsigned short sB[2][BN][LDSROW];

  const int tid = threadIdx.x;
  const int n0  = blockIdx.x * BN;
  const int m0  = blockIdx.y * BM;

  const int lane = tid & 31;
  const int wave = tid >> 5;
  const int wm = wave & 1;    // 2 wave-rows -> 64 M each
  const int wn = wave >> 1;   // 4 wave-cols -> 32 N each
  const int r  = lane & 15;
  const int hf = lane >> 4;

  // staging map: 512 16B chunks per tile -> 2 per thread
  const int rb = tid >> 2;        // 0..63
  const int cb = (tid & 3) * 8;   // element col offset

  v8f acc[4][2] = {};

  auto compute = [&](int buf) {
    Frag av[4], bv[2];
#pragma unroll
    for (int fm = 0; fm < 4; ++fm) {
      const unsigned short* p = &sA[buf][wm * 64 + fm * 16 + r][hf * 8];
      av[fm].h[0] = *(const v8bf*)p;
      av[fm].h[1] = *(const v8bf*)(p + 16);
    }
#pragma unroll
    for (int fn = 0; fn < 2; ++fn) {
      const unsigned short* p = &sB[buf][wn * 32 + fn * 16 + r][hf * 8];
      bv[fn].h[0] = *(const v8bf*)p;
      bv[fn].h[1] = *(const v8bf*)(p + 16);
    }
#pragma unroll
    for (int fm = 0; fm < 4; ++fm)
#pragma unroll
      for (int fn = 0; fn < 2; ++fn)
        acc[fm][fn] = __builtin_amdgcn_wmma_f32_16x16x32_bf16(
            false, av[fm].v, false, bv[fn].v, (short)0, acc[fm][fn],
            false, false);
  };

  const int NT = K_DIM / BK;   // 128 stages (even)

#if USE_ASYNC_LDS
  if constexpr (WBF16) {
    // pure DMA staging: no staging VGPRs, tracked by ASYNCcnt
    auto issue_stage = [&](int buf, int kt) {
      const size_t k0 = (size_t)kt * BK;
#pragma unroll
      for (int p = 0; p < 2; ++p) {
        const int row = p * 64 + rb;
        const unsigned short* ga = xnb + (size_t)(m0 + row) * K_DIM + k0 + cb;
        const unsigned short* gb = wb  + (size_t)(n0 + row) * K_DIM + k0 + cb;
        __builtin_amdgcn_global_load_async_to_lds_b128(
            AS1P(ga), AS3P(&sA[buf][row][cb]), 0, 0);
        __builtin_amdgcn_global_load_async_to_lds_b128(
            AS1P(gb), AS3P(&sB[buf][row][cb]), 0, 0);
      }
    };

    issue_stage(0, 0);
    __builtin_amdgcn_s_wait_asynccnt(0);
    __syncthreads();
    // branchless steady state: buffer indices are literals in each half
#pragma unroll 1
    for (int kt = 0; kt < NT - 2; kt += 2) {
      issue_stage(1, kt + 1);
      compute(0);
      __builtin_amdgcn_s_wait_asynccnt(0);
      __syncthreads();
      issue_stage(0, kt + 2);
      compute(1);
      __builtin_amdgcn_s_wait_asynccnt(0);
      __syncthreads();
    }
    issue_stage(1, NT - 1);
    compute(0);
    __builtin_amdgcn_s_wait_asynccnt(0);
    __syncthreads();
    compute(1);
  } else
#endif
  {
    uint4 ar[2], br[2];
    auto load_regs = [&](int kt) {
      const size_t k0 = (size_t)kt * BK;
#pragma unroll
      for (int p = 0; p < 2; ++p) {
        const int row = p * 64 + rb;
        ar[p] = *(const uint4*)(xnb + (size_t)(m0 + row) * K_DIM + k0 + cb);
        if constexpr (WBF16) {
          br[p] = *(const uint4*)(wb + (size_t)(n0 + row) * K_DIM + k0 + cb);
        } else {
          const float* wp = wq + (size_t)(n0 + row) * K_DIM + k0 + cb;
          float4 f0 = *(const float4*)wp;
          float4 f1 = *(const float4*)(wp + 4);
          br[p].x = f2bf(f0.x) | (f2bf(f0.y) << 16);
          br[p].y = f2bf(f0.z) | (f2bf(f0.w) << 16);
          br[p].z = f2bf(f1.x) | (f2bf(f1.y) << 16);
          br[p].w = f2bf(f1.z) | (f2bf(f1.w) << 16);
        }
      }
    };
    auto store_lds = [&](int buf) {
#pragma unroll
      for (int p = 0; p < 2; ++p) {
        const int row = p * 64 + rb;
        *(uint4*)&sA[buf][row][cb] = ar[p];
        *(uint4*)&sB[buf][row][cb] = br[p];
      }
    };

    load_regs(0);
    store_lds(0);
    __syncthreads();
#pragma unroll 1
    for (int kt = 0; kt < NT - 2; kt += 2) {
      load_regs(kt + 1);
      compute(0);
      store_lds(1);
      __syncthreads();
      load_regs(kt + 2);
      compute(1);
      store_lds(0);
      __syncthreads();
    }
    load_regs(NT - 1);
    compute(0);
    store_lds(1);
    __syncthreads();
    compute(1);
  }

  // Epilogue: C/D layout — VGPR q holds M = q + 8*hf, N = lane&15.
#pragma unroll
  for (int fn = 0; fn < 2; ++fn) {
    const int nc = n0 + wn * 32 + fn * 16 + r;
    const float g = gamma[nc];
#pragma unroll
    for (int fm = 0; fm < 4; ++fm) {
      const int mb = m0 + wm * 64 + fm * 16 + hf * 8;
#pragma unroll
      for (int q = 0; q < 8; ++q)
        out[(size_t)(mb + q) * N_DIM + nc] = acc[fm][fn][q] * g;
    }
  }
}

extern "C" void kernel_launch(void* const* d_in, const int* in_sizes, int n_in,
                              void* d_out, int out_size, void* d_ws, size_t ws_size,
                              hipStream_t stream) {
  const float* x     = (const float*)d_in[0];
  const float* nw    = (const float*)d_in[1];
  const float* wq    = (const float*)d_in[2];
  const float* gamma = (const float*)d_in[3];
  float* out = (float*)d_out;

  const int M = in_sizes[0] / K_DIM;                 // 8192
  const size_t xn_elems = (size_t)M * K_DIM;         // 64 MB bf16
  const size_t wb_elems = (size_t)N_DIM * K_DIM;     // 32 MB bf16

  unsigned short* xnb = (unsigned short*)d_ws;

  rmsnorm_bf16_kernel<<<M, 256, 0, stream>>>(x, nw, xnb);

  dim3 grid(N_DIM / BN, M / BM);                     // 32 x 64 blocks

  if (ws_size >= (xn_elems + wb_elems) * sizeof(unsigned short)) {
    unsigned short* wb = xnb + xn_elems;
    w_to_bf16_kernel<<<(int)(wb_elems / (256 * 8)), 256, 0, stream>>>(wq, wb);
    bitlinear_wmma_kernel<true><<<grid, 256, 0, stream>>>(xnb, wb, nullptr, gamma, out);
  } else {
    bitlinear_wmma_kernel<false><<<grid, 256, 0, stream>>>(xnb, nullptr, wq, gamma, out);
  }
}
